// LSTMCell_62938450755996
// MI455X (gfx1250) — compile-verified, matrix operand order checked by reading, not run on hardware
//
#include <hip/hip_runtime.h>
#include <hip/hip_bf16.h>
#include <stdint.h>

// ---------------------------------------------------------------------------
// LSTM cell, B=8192, I=H=1024.
// gates[8192,4096] = Acat[8192,2048] @ Wcat[4096,2048]^T  (+ bias, fused act)
//   Acat = [xt | hx] (bf16), Wcat row n: gate g=n>>10, row r=n&1023 -> [w_g | r_g]
// GEMM wave tile: 64(M) x 4 gates x 16(h) -> 16 f32 16x16 accumulators.
// ---------------------------------------------------------------------------

#define B_DIM 8192
#define I_DIM 1024
#define H_DIM 1024
#define K_DIM 2048   // I + H
#define N4H   4096   // 4 * H

typedef __attribute__((ext_vector_type(8)))  __bf16 bf16x8;
typedef __attribute__((ext_vector_type(16))) __bf16 bf16x16;
typedef __attribute__((ext_vector_type(8)))  float  f32x8;

static __device__ __forceinline__ unsigned short f32_to_bf16_rne(float f) {
    union { float f; unsigned int u; } v; v.f = f;
    unsigned int u = v.u;
    u += 0x7FFFu + ((u >> 16) & 1u);     // round-to-nearest-even
    return (unsigned short)(u >> 16);
}

// --------------------------- pack kernels ----------------------------------

__global__ __launch_bounds__(256) void pack_a(const float* __restrict__ xt,
                                              const float* __restrict__ hx,
                                              unsigned short* __restrict__ acat) {
    long flat = ((long)blockIdx.x * 256 + threadIdx.x) * 4;   // 4 elems/thread
    int row = (int)(flat >> 11);          // / 2048
    int k   = (int)(flat & 2047);
    const float* src = (k < I_DIM) ? (xt + (long)row * I_DIM + k)
                                   : (hx + (long)row * H_DIM + (k - I_DIM));
    float4 f = *(const float4*)src;
    ushort4 o;
    o.x = f32_to_bf16_rne(f.x);
    o.y = f32_to_bf16_rne(f.y);
    o.z = f32_to_bf16_rne(f.z);
    o.w = f32_to_bf16_rne(f.w);
    *(ushort4*)(acat + flat) = o;
}

__global__ __launch_bounds__(256) void pack_w(const float* __restrict__ w0, const float* __restrict__ w1,
                                              const float* __restrict__ w2, const float* __restrict__ w3,
                                              const float* __restrict__ r0, const float* __restrict__ r1,
                                              const float* __restrict__ r2, const float* __restrict__ r3,
                                              unsigned short* __restrict__ wcat) {
    long flat = ((long)blockIdx.x * 256 + threadIdx.x) * 4;
    int n = (int)(flat >> 11);            // output row 0..4095
    int k = (int)(flat & 2047);
    int g   = n >> 10;
    int row = n & 1023;
    const float* base =
        (k < I_DIM) ? (g == 0 ? w0 : g == 1 ? w1 : g == 2 ? w2 : w3)
                    : (g == 0 ? r0 : g == 1 ? r1 : g == 2 ? r2 : r3);
    long off = (long)row * 1024 + (k < I_DIM ? k : k - I_DIM);
    float4 f = *(const float4*)(base + off);
    ushort4 o;
    o.x = f32_to_bf16_rne(f.x);
    o.y = f32_to_bf16_rne(f.y);
    o.z = f32_to_bf16_rne(f.z);
    o.w = f32_to_bf16_rne(f.w);
    *(ushort4*)(wcat + flat) = o;
}

// --------------------------- WMMA GEMM + fused LSTM epilogue ---------------
//
// 16-bit A fragment layout (ISA 7.12.2): lanes 0-15 hold row M=lane with
// K chunks [0..7] (v0-3) and [16..23] (v4-7); lanes 16-31 hold the same rows
// with K chunks [8..15] and [24..31].  Both chunks are contiguous 16B in a
// K-major row, so each fragment = two global_load_b128.  B (32x16, K-major
// per column) mirrors this with N in place of M; Wcat rows are K-contiguous,
// so the identical per-lane load pattern applies.

static __device__ __forceinline__ bf16x16 load_frag(const unsigned short* p) {
    bf16x8 c0 = *(const bf16x8*)(p);        // K 0..7   (or 8..15 on hi lanes)
    bf16x8 c1 = *(const bf16x8*)(p + 16);   // K 16..23 (or 24..31)
    return __builtin_shufflevector(c0, c1,
        0, 1, 2, 3, 4, 5, 6, 7, 8, 9, 10, 11, 12, 13, 14, 15);
}

static __device__ __forceinline__ float sigmoid_f(float x) {
    return 1.0f / (1.0f + __expf(-x));
}
static __device__ __forceinline__ float tanh_f(float x) {
    return 2.0f / (1.0f + __expf(-2.0f * x)) - 1.0f;
}

#define WMMA_BF16(A, Bm, C) \
    __builtin_amdgcn_wmma_f32_16x16x32_bf16(false, (A), false, (Bm), (short)0, (C), false, false)

__global__ __launch_bounds__(256) void lstm_gemm(
    const unsigned short* __restrict__ acat,   // [8192, 2048] bf16
    const unsigned short* __restrict__ wcat,   // [4096, 2048] bf16
    const float* __restrict__ b_fg, const float* __restrict__ b_ig,
    const float* __restrict__ b_og, const float* __restrict__ b_ci,
    const float* __restrict__ cx,              // [8192, 1024]
    float* __restrict__ ht, float* __restrict__ ct)
{
    const int lane = threadIdx.x & 31;
    const int wave = threadIdx.x >> 5;    // 8 waves: 4 in M x 2 in N
    const int wm   = wave & 3;
    const int wn   = wave >> 2;
    const int lo   = lane & 15;
    const int hi   = lane >> 4;

    const int m0 = blockIdx.x * 256 + wm * 64;   // 64 rows per wave (4 subtiles)
    const int h0 = blockIdx.y * 32  + wn * 16;   // 16 h-columns per wave

    f32x8 acc[4][4];                             // [m-subtile][gate]
    #pragma unroll
    for (int s = 0; s < 4; ++s)
        #pragma unroll
        for (int g = 0; g < 4; ++g)
            #pragma unroll
            for (int e = 0; e < 8; ++e)
                acc[s][g][e] = 0.0f;

    // per-lane fragment base pointers (16B aligned: hi*8 elems = 16B)
    const unsigned short* aptr0 = acat + (long)(m0 +  0 + lo) * K_DIM + hi * 8;
    const unsigned short* aptr1 = acat + (long)(m0 + 16 + lo) * K_DIM + hi * 8;
    const unsigned short* aptr2 = acat + (long)(m0 + 32 + lo) * K_DIM + hi * 8;
    const unsigned short* aptr3 = acat + (long)(m0 + 48 + lo) * K_DIM + hi * 8;
    const unsigned short* bptr0 = wcat + (long)(0 * H_DIM + h0 + lo) * K_DIM + hi * 8;
    const unsigned short* bptr1 = wcat + (long)(1 * H_DIM + h0 + lo) * K_DIM + hi * 8;
    const unsigned short* bptr2 = wcat + (long)(2 * H_DIM + h0 + lo) * K_DIM + hi * 8;
    const unsigned short* bptr3 = wcat + (long)(3 * H_DIM + h0 + lo) * K_DIM + hi * 8;

    #pragma unroll 2
    for (int k0 = 0; k0 < K_DIM; k0 += 32) {
        bf16x16 a0 = load_frag(aptr0 + k0);
        bf16x16 a1 = load_frag(aptr1 + k0);
        bf16x16 a2 = load_frag(aptr2 + k0);
        bf16x16 a3 = load_frag(aptr3 + k0);

        bf16x16 b0 = load_frag(bptr0 + k0);
        acc[0][0] = WMMA_BF16(a0, b0, acc[0][0]);
        acc[1][0] = WMMA_BF16(a1, b0, acc[1][0]);
        acc[2][0] = WMMA_BF16(a2, b0, acc[2][0]);
        acc[3][0] = WMMA_BF16(a3, b0, acc[3][0]);

        bf16x16 b1 = load_frag(bptr1 + k0);
        acc[0][1] = WMMA_BF16(a0, b1, acc[0][1]);
        acc[1][1] = WMMA_BF16(a1, b1, acc[1][1]);
        acc[2][1] = WMMA_BF16(a2, b1, acc[2][1]);
        acc[3][1] = WMMA_BF16(a3, b1, acc[3][1]);

        bf16x16 b2 = load_frag(bptr2 + k0);
        acc[0][2] = WMMA_BF16(a0, b2, acc[0][2]);
        acc[1][2] = WMMA_BF16(a1, b2, acc[1][2]);
        acc[2][2] = WMMA_BF16(a2, b2, acc[2][2]);
        acc[3][2] = WMMA_BF16(a3, b2, acc[3][2]);

        bf16x16 b3 = load_frag(bptr3 + k0);
        acc[0][3] = WMMA_BF16(a0, b3, acc[0][3]);
        acc[1][3] = WMMA_BF16(a1, b3, acc[1][3]);
        acc[2][3] = WMMA_BF16(a2, b3, acc[2][3]);
        acc[3][3] = WMMA_BF16(a3, b3, acc[3][3]);
    }

    // ---- fused LSTM epilogue -------------------------------------------
    // C/D layout: lane 0-15 -> N=lane, vgpr r -> M=r; lane 16-31 -> M=r+8.
    const int h = h0 + lo;
    const float bf = b_fg[h];
    const float bi = b_ig[h];
    const float bo = b_og[h];
    const float bg = b_ci[h];

    #pragma unroll
    for (int s = 0; s < 4; ++s) {
        #pragma unroll
        for (int r = 0; r < 8; ++r) {
            const int m = m0 + s * 16 + hi * 8 + r;
            const long idx = (long)m * H_DIM + h;
            float fv = acc[s][0][r] + bf;
            float iv = acc[s][1][r] + bi;
            float ov = acc[s][2][r] + bo;
            float gv = acc[s][3][r] + bg;
            float ftv = sigmoid_f(fv);
            float itv = sigmoid_f(iv);
            float otv = sigmoid_f(ov);
            float gtv = tanh_f(gv);
            float cv  = ftv * cx[idx] + itv * gtv;
            float hv  = otv * tanh_f(cv);
            ht[idx] = hv;
            ct[idx] = cv;
        }
    }
}

// --------------------------- launcher --------------------------------------

extern "C" void kernel_launch(void* const* d_in, const int* in_sizes, int n_in,
                              void* d_out, int out_size, void* d_ws, size_t ws_size,
                              hipStream_t stream) {
    const float* xt = (const float*)d_in[0];
    const float* cx = (const float*)d_in[1];
    const float* hx = (const float*)d_in[2];
    const float* w_fg = (const float*)d_in[3];
    const float* w_ig = (const float*)d_in[4];
    const float* w_og = (const float*)d_in[5];
    const float* w_ci = (const float*)d_in[6];
    const float* r_fg = (const float*)d_in[7];
    const float* r_ig = (const float*)d_in[8];
    const float* r_og = (const float*)d_in[9];
    const float* r_ci = (const float*)d_in[10];
    const float* b_fg = (const float*)d_in[11];
    const float* b_ig = (const float*)d_in[12];
    const float* b_og = (const float*)d_in[13];
    const float* b_ci = (const float*)d_in[14];

    unsigned short* acat = (unsigned short*)d_ws;                    // 32 MB
    unsigned short* wcat = acat + (size_t)B_DIM * K_DIM;             // +16 MB

    float* ht = (float*)d_out;
    float* ct = ht + (size_t)B_DIM * H_DIM;

    {   // Acat: 8192*2048 bf16, 4 elems/thread
        const long n_thr = (long)B_DIM * K_DIM / 4;
        pack_a<<<(unsigned)(n_thr / 256), 256, 0, stream>>>(xt, hx, acat);
    }
    {   // Wcat: 4096*2048 bf16, 4 elems/thread
        const long n_thr = (long)N4H * K_DIM / 4;
        pack_w<<<(unsigned)(n_thr / 256), 256, 0, stream>>>(
            w_fg, w_ig, w_og, w_ci, r_fg, r_ig, r_og, r_ci, wcat);
    }
    {
        dim3 grid(B_DIM / 256, H_DIM / 32);   // (32, 32)
        lstm_gemm<<<grid, 256, 0, stream>>>(acat, wcat,
                                            b_fg, b_ig, b_og, b_ci,
                                            cx, ht, ct);
    }
}
